// SceneGraphGNN_72851235275082
// MI455X (gfx1250) — matrix operand order repaired and checked by reading, not compiled, not run on hardware
//
#include <hip/hip_runtime.h>

// ---------------------------------------------------------------------------
// SceneGraph GNN forward for MI455X (gfx1250).
// Dense matmuls -> v_wmma_f32_16x16x32_bf16 (bf16 in, f32 accumulate).
// Per-wave tile 16x128 (8 f32 accumulators) + staged B fragments so loads
// issue as one clause and WMMAs overlap outstanding loads.
// Sparse scatter -> L2-resident global_atomic_add_f32 (hw buffer 51MB << 192MB L2).
// ---------------------------------------------------------------------------

typedef __attribute__((ext_vector_type(16))) __bf16 v16bf;
typedef __attribute__((ext_vector_type(8)))  float  v8f;

#define H_DIM 256

union FragBF {
    v16bf v;
    uint4 q[2];
};

__device__ __forceinline__ __bf16 f2bf(float f) { return (__bf16)f; }

// ---- conversion / layout kernels ------------------------------------------

// f32 [rows, cols] row-major -> bf16 [rows, colsPad] row-major, zero padded
__global__ void cvt_pad_bf16(const float* __restrict__ src, __bf16* __restrict__ dst,
                             int rows, int cols, int colsPad) {
    long long i = (long long)blockIdx.x * blockDim.x + threadIdx.x;
    long long total = (long long)rows * colsPad;
    if (i >= total) return;
    int r = (int)(i / colsPad);
    int c = (int)(i % colsPad);
    dst[i] = (c < cols) ? f2bf(src[(long long)r * cols + c]) : f2bf(0.0f);
}

// W f32 [K, Nc] row-major -> Wt bf16 [Nc, Kpad] row-major (n-major), zero pad K
__global__ void w_transpose_bf16(const float* __restrict__ W, __bf16* __restrict__ Wt,
                                 int K, int Nc, int Kpad) {
    long long i = (long long)blockIdx.x * blockDim.x + threadIdx.x;
    long long total = (long long)Nc * Kpad;
    if (i >= total) return;
    int n = (int)(i / Kpad);
    int k = (int)(i % Kpad);
    Wt[i] = (k < K) ? f2bf(W[(long long)k * Nc + n]) : f2bf(0.0f);
}

// ---- degree / normalization -----------------------------------------------

__global__ void deg_init(float* __restrict__ deg, int n) {
    int i = blockIdx.x * blockDim.x + threadIdx.x;
    if (i < n) deg[i] = 1.0f;  // self-loop
}

__global__ void deg_count(const int* __restrict__ dst, float* __restrict__ deg, int e) {
    int i = blockIdx.x * blockDim.x + threadIdx.x;
    if (i < e) atomicAdd(&deg[dst[i]], 1.0f);
}

__global__ void deg_to_dis(float* __restrict__ deg, int n) {
    int i = blockIdx.x * blockDim.x + threadIdx.x;
    if (i < n) deg[i] = rsqrtf(deg[i]);  // deg^-0.5, in place
}

// ---- WMMA GEMM: out[M,256] = A[M,K](bf16) x Wt[256,K](bf16, n-major) ------
// MODE 0: outb = bf16(relu(acc + bias))            (encoder layer 1)
// MODE 1: outb = bf16(acc + bias)                  (encoder layer 2)
// MODE 2: hwf  = acc (f32);  cinit = acc*dis[r]^2 + bias   (GCN layers)
//
// K is a template constant (544 / 256) so B-tile offsets (nt*16*K elements)
// fold into global_load immediate offsets; base pointers step by 64B per k.
//
// Block = 8 waves = 4 row-tiles x 2 col-halves: 64 rows x 256 cols per block.
// Wave tile = 16 rows x 128 cols = 8 accumulators. A-frag + all 8 B-frags are
// staged into registers first (one load clause), then the 8-WMMA chain runs
// with staggered load waits.
//
// A-frag (16x32 bf16): lane&15 = M-row; (lane>>4) selects K bytes {0..7,16..23}
// vs {8..15,24..31}. B-frag (32x16 bf16): lane&15 = N-col; same K split as A
// ({0..7,16..23} vs {8..15,24..31}) (hence the n-major Wt layout).
template <int MODE, int K>
__global__ __launch_bounds__(256, 1)
void gemm_bf16_wmma(const __bf16* __restrict__ A, const __bf16* __restrict__ Wt,
                    const float* __restrict__ bias, const float* __restrict__ dis,
                    __bf16* __restrict__ outb, float* __restrict__ hwf,
                    float* __restrict__ cinit, int M) {
    const int lane    = threadIdx.x & 31;
    const int wave    = threadIdx.x >> 5;
    const int waveRow = wave >> 1;                 // 0..3
    const int waveCol = wave & 1;                  // 0..1
    const int mtile   = blockIdx.x * 4 + waveRow;  // 16-row tile id
    if (mtile * 16 >= M) return;

    int row = mtile * 16 + (lane & 15);
    if (row >= M) row = M - 1;                     // clamped loads, guarded stores
    const int kbA = (lane >> 4) * 8;               // A sub-K offset {0,8}
    const int kbB = (lane >> 4) * 8;               // B sub-K offset {0,8}
    const int n0  = waveCol * 128;                 // 8 col tiles of 16

    v8f zero = {0.f, 0.f, 0.f, 0.f, 0.f, 0.f, 0.f, 0.f};
    v8f c[8];
#pragma unroll
    for (int i = 0; i < 8; ++i) c[i] = zero;

    const __bf16* ap  = A + (long long)row * K + kbA;
    const __bf16* bp0 = Wt + (long long)(n0 + (lane & 15)) * K + kbB;

    for (int k0 = 0; k0 < K; k0 += 32) {
        // stage A fragment + all 8 B fragments (one load clause, 18x b128)
        FragBF a;
        a.q[0] = *(const uint4*)(ap);
        a.q[1] = *(const uint4*)(ap + 16);
        FragBF b[8];
#pragma unroll
        for (int nt = 0; nt < 8; ++nt) {
            b[nt].q[0] = *(const uint4*)(bp0 + nt * 16 * K);      // const offsets
            b[nt].q[1] = *(const uint4*)(bp0 + nt * 16 * K + 16);
        }
        // 8-WMMA chain; waits stagger as loads drain
#pragma unroll
        for (int nt = 0; nt < 8; ++nt) {
            c[nt] = __builtin_amdgcn_wmma_f32_16x16x32_bf16(
                false, a.v, false, b[nt].v, (short)0, c[nt], false, false);
        }
        ap  += 32;
        bp0 += 32;
    }

    // D layout: element v of acc holds M = mtile*16 + (lane>>4)*8 + v,
    //           N = n0 + nt*16 + (lane&15)
    const int rbase = mtile * 16 + (lane >> 4) * 8;
    const int ncol0 = n0 + (lane & 15);
#pragma unroll
    for (int nt = 0; nt < 8; ++nt) {
        const int n  = ncol0 + nt * 16;
        const float bn = bias[n];
#pragma unroll
        for (int v = 0; v < 8; ++v) {
            const int r = rbase + v;
            if (r >= M) continue;
            const float acc = c[nt][v];
            if (MODE == 0) {
                float val = acc + bn;
                outb[(long long)r * H_DIM + n] = f2bf(val > 0.f ? val : 0.f);
            } else if (MODE == 1) {
                outb[(long long)r * H_DIM + n] = f2bf(acc + bn);
            } else {
                hwf[(long long)r * H_DIM + n] = acc;
                const float dn = dis[r];
                cinit[(long long)r * H_DIM + n] = acc * dn * dn + bn;
            }
        }
    }
}

// ---- edge scatter: C[dst] += hw[src] * dis[src]*dis[dst] ------------------
// One wave per edge, 8 coalesced f32 atomics per lane (256 cols).
__global__ void edge_scatter(const int* __restrict__ src, const int* __restrict__ dst,
                             const float* __restrict__ dis,
                             const float* __restrict__ hwf, float* __restrict__ C, int E) {
    const int e = blockIdx.x * 8 + (threadIdx.x >> 5);
    if (e >= E) return;
    const int lane = threadIdx.x & 31;
    const int s = src[e];
    const int d = dst[e];
    const float w = dis[s] * dis[d];
    const float* hs = hwf + (long long)s * H_DIM;
    float* cd = C + (long long)d * H_DIM;
#pragma unroll
    for (int t = 0; t < 8; ++t) {
        const int j = lane + t * 32;
        atomicAdd(cd + j, hs[j] * w);
    }
}

// ---- relu(C) -> bf16 for next layer ---------------------------------------
__global__ void relu_to_bf16(const float* __restrict__ C, __bf16* __restrict__ HB,
                             long long total) {
    long long i = (long long)blockIdx.x * blockDim.x + threadIdx.x;
    if (i >= total) return;
    float v = C[i];
    HB[i] = f2bf(v > 0.f ? v : 0.f);
}

// ---- column sums of relu(C) for mean pooling ------------------------------
__global__ void zero_sums(float* __restrict__ sums) { sums[threadIdx.x] = 0.f; }

__global__ void col_sum_relu(const float* __restrict__ C, float* __restrict__ sums, int M) {
    const int col = threadIdx.x;                  // 256 threads = 256 cols
    int r0 = blockIdx.x * 128;
    int r1 = r0 + 128;
    if (r1 > M) r1 = M;
    float acc = 0.f;
    for (int r = r0; r < r1; ++r) {
        const float v = C[(long long)r * H_DIM + col];
        acc += (v > 0.f) ? v : 0.f;
    }
    atomicAdd(&sums[col], acc);
}

// ---- head MLP: out[128] = relu(mean@W1+b1)@W2+b2 --------------------------
__global__ void head_mlp(const float* __restrict__ sums, float invN,
                         const float* __restrict__ W1, const float* __restrict__ b1,
                         const float* __restrict__ W2, const float* __restrict__ b2,
                         float* __restrict__ out) {
    __shared__ float meanv[256];
    __shared__ float z[128];
    const int t = threadIdx.x;  // 256 threads
    meanv[t] = sums[t] * invN;
    __syncthreads();
    if (t < 128) {
        float a = b1[t];
        for (int k = 0; k < 256; ++k) a += meanv[k] * W1[k * 128 + t];
        z[t] = a > 0.f ? a : 0.f;
    }
    __syncthreads();
    if (t < 128) {
        float a = b2[t];
        for (int k = 0; k < 128; ++k) a += z[k] * W2[k * 128 + t];
        out[t] = a;
    }
}

// ---------------------------------------------------------------------------

extern "C" void kernel_launch(void* const* d_in, const int* in_sizes, int n_in,
                              void* d_out, int out_size, void* d_ws, size_t ws_size,
                              hipStream_t stream) {
    const float* x      = (const float*)d_in[0];
    const int*   eidx   = (const int*)d_in[1];
    // d_in[2] edge_attr, d_in[7] emb_table: unused downstream (matches reference)
    const float* W_enc1 = (const float*)d_in[3];
    const float* b_enc1 = (const float*)d_in[4];
    const float* W_enc2 = (const float*)d_in[5];
    const float* b_enc2 = (const float*)d_in[6];
    const float* W_g1   = (const float*)d_in[8];
    const float* b_g1   = (const float*)d_in[9];
    const float* W_g2   = (const float*)d_in[10];
    const float* b_g2   = (const float*)d_in[11];
    const float* W_g3   = (const float*)d_in[12];
    const float* b_g3   = (const float*)d_in[13];
    const float* W_h1   = (const float*)d_in[14];
    const float* b_h1   = (const float*)d_in[15];
    const float* W_h2   = (const float*)d_in[16];
    const float* b_h2   = (const float*)d_in[17];

    const int H    = in_sizes[4];             // 256
    const int D_IN = in_sizes[3] / H;         // 518
    const int N    = in_sizes[0] / D_IN;      // 50000
    const int E    = in_sizes[1] / 2;         // 1.6M
    const int KP1  = ((D_IN + 31) / 32) * 32; // 544 (compile-time in GEMM template)

    const int* src = eidx;
    const int* dst = eidx + E;

    // workspace carving (256B aligned)
    char* ws = (char*)d_ws;
    size_t off = 0;
    auto alloc = [&](size_t bytes) -> void* {
        void* p = ws + off;
        off += (bytes + 255) & ~(size_t)255;
        return p;
    };
    __bf16* xb   = (__bf16*)alloc((size_t)N * KP1 * 2);   // padded bf16 x
    __bf16* hba  = (__bf16*)alloc((size_t)N * H * 2);     // ping
    __bf16* hbb  = (__bf16*)alloc((size_t)N * H * 2);     // pong
    float*  hwf  = (float*)alloc((size_t)N * H * 4);      // h@W (f32, gather src)
    float*  Cbuf = (float*)alloc((size_t)N * H * 4);      // scatter accumulator
    float*  dis  = (float*)alloc((size_t)N * 4);          // deg^-0.5
    __bf16* wt1  = (__bf16*)alloc((size_t)H * KP1 * 2);
    __bf16* wt2  = (__bf16*)alloc((size_t)H * H * 2);
    __bf16* wg1  = (__bf16*)alloc((size_t)H * H * 2);
    __bf16* wg2  = (__bf16*)alloc((size_t)H * H * 2);
    __bf16* wg3  = (__bf16*)alloc((size_t)H * H * 2);
    float*  sums = (float*)alloc(256 * 4);
    (void)ws_size; (void)n_in; (void)out_size;

    const dim3 blk(256);

    // 1) convert inputs / weights to WMMA layouts
    {
        long long total = (long long)N * KP1;
        cvt_pad_bf16<<<(unsigned)((total + 255) / 256), blk, 0, stream>>>(x, xb, N, D_IN, KP1);
    }
    {
        long long t1 = (long long)H * KP1;
        w_transpose_bf16<<<(unsigned)((t1 + 255) / 256), blk, 0, stream>>>(W_enc1, wt1, D_IN, H, KP1);
        long long t2 = (long long)H * H;
        unsigned g2 = (unsigned)((t2 + 255) / 256);
        w_transpose_bf16<<<g2, blk, 0, stream>>>(W_enc2, wt2, H, H, H);
        w_transpose_bf16<<<g2, blk, 0, stream>>>(W_g1, wg1, H, H, H);
        w_transpose_bf16<<<g2, blk, 0, stream>>>(W_g2, wg2, H, H, H);
        w_transpose_bf16<<<g2, blk, 0, stream>>>(W_g3, wg3, H, H, H);
    }

    // 2) degree -> dis = deg^-0.5 (self-loop included)
    deg_init<<<(N + 255) / 256, blk, 0, stream>>>(dis, N);
    deg_count<<<(E + 255) / 256, blk, 0, stream>>>(dst, dis, E);
    deg_to_dis<<<(N + 255) / 256, blk, 0, stream>>>(dis, N);

    const unsigned gemmGrid = (unsigned)((N + 63) / 64);  // 4 row tiles x 16 rows / block

    // 3) encoder: h = relu(x@W1+b1)@W2+b2   (bf16 activations)
    gemm_bf16_wmma<0, 544><<<gemmGrid, blk, 0, stream>>>(xb, wt1, b_enc1, nullptr,
                                                         hba, nullptr, nullptr, N);
    gemm_bf16_wmma<1, 256><<<gemmGrid, blk, 0, stream>>>(hba, wt2, b_enc2, nullptr,
                                                         hbb, nullptr, nullptr, N);

    const unsigned scatGrid = (unsigned)((E + 7) / 8);
    const long long nh = (long long)N * H;
    const unsigned reluGrid = (unsigned)((nh + 255) / 256);

    // 4) GCN layer 1: hbb -> hba
    gemm_bf16_wmma<2, 256><<<gemmGrid, blk, 0, stream>>>(hbb, wg1, b_g1, dis,
                                                         nullptr, hwf, Cbuf, N);
    edge_scatter<<<scatGrid, blk, 0, stream>>>(src, dst, dis, hwf, Cbuf, E);
    relu_to_bf16<<<reluGrid, blk, 0, stream>>>(Cbuf, hba, nh);

    // 5) GCN layer 2: hba -> hbb
    gemm_bf16_wmma<2, 256><<<gemmGrid, blk, 0, stream>>>(hba, wg2, b_g2, dis,
                                                         nullptr, hwf, Cbuf, N);
    edge_scatter<<<scatGrid, blk, 0, stream>>>(src, dst, dis, hwf, Cbuf, E);
    relu_to_bf16<<<reluGrid, blk, 0, stream>>>(Cbuf, hbb, nh);

    // 6) GCN layer 3: hbb -> Cbuf (relu fused into reduction)
    gemm_bf16_wmma<2, 256><<<gemmGrid, blk, 0, stream>>>(hbb, wg3, b_g3, dis,
                                                         nullptr, hwf, Cbuf, N);
    edge_scatter<<<scatGrid, blk, 0, stream>>>(src, dst, dis, hwf, Cbuf, E);

    // 7) mean pool + head MLP
    zero_sums<<<1, blk, 0, stream>>>(sums);
    col_sum_relu<<<(unsigned)((N + 127) / 128), blk, 0, stream>>>(Cbuf, sums, N);
    head_mlp<<<1, blk, 0, stream>>>(sums, 1.0f / (float)N, W_h1, b_h1, W_h2, b_h2,
                                    (float*)d_out);
}
